// DCNTranspose_55207509623211
// MI455X (gfx1250) — compile-verified, matrix operand order checked by reading, not run on hardware
//
#include <hip/hip_runtime.h>
#include <hip/hip_bf16.h>
#include <math.h>

typedef __attribute__((ext_vector_type(16))) _Float16 v16h;
typedef __attribute__((ext_vector_type(8)))  _Float16 v8h;
typedef __attribute__((ext_vector_type(8)))  float    v8f;

// Problem constants
#define Bn   4
#define CIN  64
#define HIN  64
#define WIN  64
#define COUT 64
#define HO   128
#define WO   128
#define KTAP 16
#define NPIX (Bn*HO*WO)          // 65536
#define APITCH 40                // padded halfs per tile row (80B = 20 words)

// ---------------------------------------------------------------------------
// Pack w_dcn (flipped) into fragment-ready chunk order:
//   wAp[(k*2+half)*2048 + m*32 + cc] = f16(w_dcn[m, half*32+cc, 3-ka, 3-kb])
// so each K-chunk (k,half) is a contiguous 4KB [m][cc] block.
// ---------------------------------------------------------------------------
__global__ __launch_bounds__(256) void pack_wdcn_kernel(const float* __restrict__ w_dcn,
                                                        _Float16* __restrict__ wAp) {
    int idx = blockIdx.x * 256 + threadIdx.x;          // 65536
    if (idx >= KTAP * CIN * COUT) return;
    int cc   = idx & 31;
    int m    = (idx >> 5) & 63;
    int half = (idx >> 11) & 1;
    int k    = idx >> 12;
    int c  = half * 32 + cc;
    int ka = k >> 2, kb = k & 3;
    float v = w_dcn[m * (CIN * 16) + c * 16 + (3 - ka) * 4 + (3 - kb)];
    wAp[idx] = (_Float16)v;
}

// ---------------------------------------------------------------------------
// Pack w_om into wOMp[ky*4+kx][c][ch]: 48 channels contiguous for wide s_loads
// ---------------------------------------------------------------------------
__global__ __launch_bounds__(256) void pack_wom_kernel(const float* __restrict__ w_om,
                                                       float* __restrict__ wOMp) {
    int idx = blockIdx.x * 256 + threadIdx.x;          // 49152
    if (idx >= 16 * CIN * 48) return;
    int tap = idx / (CIN * 48);
    int c   = (idx / 48) % CIN;
    int ch  = idx % 48;
    wOMp[idx] = w_om[ch * (CIN * 16) + c * 16 + tap];
}

// ---------------------------------------------------------------------------
// Stage 1: offset/mask conv (zero-stuffing -> only 2x2 live taps), softmax,
// then collapse bilinear-on-zero-stuffed-grid to one (src, wgt) per pixel/tap.
// ---------------------------------------------------------------------------
__global__ __launch_bounds__(64) void dcn_offset_kernel(const float* __restrict__ x,
                                                        const float* __restrict__ wOMp,
                                                        const float* __restrict__ b_om,
                                                        float* __restrict__ wgtArr,
                                                        int*   __restrict__ srcArr) {
    const int ho = blockIdx.x;
    const int b  = blockIdx.y;
    const int t  = threadIdx.x;                        // 0..63

    __shared__ float xs[2][CIN][WIN];                  // two x rows, 32 KB

    const int  kyb = (ho & 1);
    const int  y0  = ho + kyb, y1 = y0 + 2;
    const bool rv0 = (y0 >= 2) && (y0 <= 128);
    const bool rv1 = (y1 >= 2) && (y1 <= 128);
    const int  r0  = (y0 - 2) >> 1, r1 = (y1 - 2) >> 1;

    const float* xb = x + (size_t)b * CIN * HIN * WIN;
    for (int idx = t; idx < 2 * CIN * WIN; idx += 64) {
        int i = idx >> 12, rem = idx & 4095;
        int c = rem >> 6, col = rem & 63;
        bool rv = i ? rv1 : rv0;
        int  r  = i ? r1 : r0;
        xs[i][c][col] = rv ? xb[c * (HIN * WIN) + r * WIN + col] : 0.0f;
    }
    __syncthreads();

    float acc[2][48];
#pragma unroll
    for (int p = 0; p < 2; ++p)
#pragma unroll
        for (int ch = 0; ch < 48; ++ch) acc[p][ch] = 0.0f;

    const int w0 = 2 * t;
#pragma unroll
    for (int iy = 0; iy < 2; ++iy) {
        bool rv = iy ? rv1 : rv0;
        if (rv) {
            int ky = kyb + 2 * iy;
#pragma unroll
            for (int kx = 0; kx < 4; ++kx) {
                const int pix = kx & 1;
                const int wp  = w0 + pix;
                const int xcoord = wp + kx;
                const bool cv = (xcoord >= 2) && (xcoord <= 128);
                const int  col = cv ? ((xcoord - 2) >> 1) : 0;
                const float* wbase = wOMp + (size_t)((ky * 4 + kx) * CIN) * 48;
                for (int c = 0; c < CIN; ++c) {
                    float xv = cv ? xs[iy][c][col] : 0.0f;
                    const float* wr = wbase + c * 48;  // uniform -> s_load
#pragma unroll
                    for (int ch = 0; ch < 48; ++ch)
                        acc[pix][ch] = fmaf(wr[ch], xv, acc[pix][ch]);
                }
            }
        }
    }

    const int p_base = (b * (HO * WO) + ho * WO);
#pragma unroll
    for (int pix = 0; pix < 2; ++pix) {
        const int wo = w0 + pix;
        float om[48];
#pragma unroll
        for (int ch = 0; ch < 48; ++ch) om[ch] = acc[pix][ch] + b_om[ch];

        float mx = om[32];
#pragma unroll
        for (int k = 1; k < KTAP; ++k) mx = fmaxf(mx, om[32 + k]);
        float e[KTAP]; float s = 0.0f;
#pragma unroll
        for (int k = 0; k < KTAP; ++k) { e[k] = __expf(om[32 + k] - mx); s += e[k]; }
        const float inv = 1.0f / s;

#pragma unroll
        for (int k = 0; k < KTAP; ++k) {
            const float dy = om[2 * k], dx = om[2 * k + 1];
            const int ka = k >> 2, kb = k & 3;
            const float py = dy + (float)(ka + ho);
            const float px = dx + (float)(kb + wo);
            const float fy = floorf(py), fx = floorf(px);
            const int y0i = (int)fy, x0i = (int)fx;
            const float ly = py - fy, lx = px - fx;
            int ye, xe; float wy, wx;
            if ((y0i & 1) == 0) { ye = y0i;     wy = 1.0f - ly; }
            else               { ye = y0i + 1; wy = ly; }
            if ((x0i & 1) == 0) { xe = x0i;     wx = 1.0f - lx; }
            else               { xe = x0i + 1; wx = lx; }
            const bool val = (ye >= 2) && (ye <= 128) && (xe >= 2) && (xe <= 128);
            const float wgt = val ? (wy * wx * e[k] * inv) : 0.0f;
            const int   src = val ? (((ye - 2) >> 1) * WIN + ((xe - 2) >> 1)) : 0;
            const int pp = p_base + wo;
            wgtArr[pp * KTAP + k] = wgt;
            srcArr[pp * KTAP + k] = src;
        }
    }
}

// ---------------------------------------------------------------------------
// Stage 2: gather-GEMM with WMMA.
//   A[m, k*64+c] = f16(w_flip[m,c,k])   (async-copied to LDS, fragment-ready)
//   B[k*64+c, n] = f16(wgt[n,k] * x[b,c,src[n,k]])
// Tiles stored [row][cc] with 40-half pitch -> each v16h fragment is two
// ds_load_b128; B staging packs 8 halfs per ds_store_b128.
// ---------------------------------------------------------------------------
__global__ __launch_bounds__(256) void dcn_gemm_kernel(const float* __restrict__ x,
                                                       const _Float16* __restrict__ wAp,
                                                       const float* __restrict__ b_dcn,
                                                       const float* __restrict__ wgtArr,
                                                       const int*   __restrict__ srcArr,
                                                       float* __restrict__ out) {
    __shared__ float               sWgt[KTAP * 64];        // [k][n]
    __shared__ int                 sSrc[KTAP * 64];        // [k][n]
    __shared__ __align__(16) _Float16 sA[COUT * APITCH];   // [m][cc], padded
    __shared__ __align__(16) _Float16 sB[64 * APITCH];     // [n][cc], padded

    const int t   = threadIdx.x;
    const int p0  = blockIdx.x * 64;
    const int b   = p0 >> 14;
    const int ho  = (p0 >> 7) & 127;
    const int wo0 = p0 & 127;

    for (int i = t; i < KTAP * 64; i += 256) {
        int n = i >> 4, k = i & 15;
        sWgt[k * 64 + n] = wgtArr[p0 * KTAP + i];
        sSrc[k * 64 + n] = srcArr[p0 * KTAP + i];
    }

    const int lane  = t & 31;
    const int w     = t >> 5;
    const int msub  = (w & 3) << 4;
    const int nsub0 = (w >> 2) << 4;
    const int lm    = lane & 15;
    const int h     = lane >> 4;

    // per-thread async A-copy assignment: 16B of row am, segment aseg
    const int am   = t >> 2;
    const int aseg = t & 3;
    const uint32_t ldsA = (uint32_t)(uintptr_t)(sA + am * APITCH + aseg * 8);

    // per-thread B staging assignment
    const int bn = t & 63;
    const int bg = t >> 6;

    v8f acc0 = {}; v8f acc1 = {};
    const float* xb = x + (size_t)b * (CIN * HIN * WIN);

    __syncthreads();

    for (int q = 0; q < 32; ++q) {
        const int k  = q >> 1;
        const int c0 = (q & 1) << 5;

        // --- async A tile: contiguous 4KB chunk -> LDS, no VGPR round trip
        {
            uint64_t ga = (uint64_t)(uintptr_t)(wAp + (size_t)q * 2048 + (size_t)t * 8);
            asm volatile("global_load_async_to_lds_b128 %0, %1, off"
                         :: "v"(ldsA), "v"(ga) : "memory");
        }

        // --- B tile: gather + scale + convert, one b128 store per thread
        {
            const float wv = sWgt[k * 64 + bn];
            const int   si = sSrc[k * 64 + bn];
            const float* xc = xb + si + (size_t)(c0 + bg * 8) * (HIN * WIN);
            v8h hb;
#pragma unroll
            for (int i = 0; i < 8; ++i)
                hb[i] = (_Float16)(wv * xc[(size_t)i * (HIN * WIN)]);
            *(v8h*)(sB + bn * APITCH + bg * 8) = hb;
        }

        asm volatile("s_wait_asynccnt 0x0" ::: "memory");
        __syncthreads();

        // --- fragments: two ds_load_b128 each, per the 16-bit wave32 layout
        const _Float16* arow  = sA + (msub + lm) * APITCH + 8 * h;
        const _Float16* brow0 = sB + (nsub0 + lm) * APITCH + 8 * h;
        const _Float16* brow1 = sB + (nsub0 + 32 + lm) * APITCH + 8 * h;
        v8h alo = *(const v8h*)(arow);
        v8h ahi = *(const v8h*)(arow + 16);
        v8h blo0 = *(const v8h*)(brow0);
        v8h bhi0 = *(const v8h*)(brow0 + 16);
        v8h blo1 = *(const v8h*)(brow1);
        v8h bhi1 = *(const v8h*)(brow1 + 16);
        v16h a  = __builtin_shufflevector(alo,  ahi,  0,1,2,3,4,5,6,7,8,9,10,11,12,13,14,15);
        v16h bf0 = __builtin_shufflevector(blo0, bhi0, 0,1,2,3,4,5,6,7,8,9,10,11,12,13,14,15);
        v16h bf1 = __builtin_shufflevector(blo1, bhi1, 0,1,2,3,4,5,6,7,8,9,10,11,12,13,14,15);

        acc0 = __builtin_amdgcn_wmma_f32_16x16x32_f16(false, a, false, bf0,
                                                      (short)0, acc0, false, false);
        acc1 = __builtin_amdgcn_wmma_f32_16x16x32_f16(false, a, false, bf1,
                                                      (short)0, acc1, false, false);
        __syncthreads();
    }

    // epilogue: bias + store (D: VGPR r, lane L -> M = r + 8*(L/16), N = L%16)
#pragma unroll
    for (int r = 0; r < 8; ++r) {
        const int m = msub + r + 8 * h;
        const float bias = b_dcn[m];
        const size_t base = (((size_t)b * COUT + m) * HO + ho) * WO + wo0;
        out[base + nsub0 + lm]      = acc0[r] + bias;
        out[base + nsub0 + 32 + lm] = acc1[r] + bias;
    }
}

// ---------------------------------------------------------------------------
extern "C" void kernel_launch(void* const* d_in, const int* in_sizes, int n_in,
                              void* d_out, int out_size, void* d_ws, size_t ws_size,
                              hipStream_t stream) {
    const float* x     = (const float*)d_in[0];
    const float* w_om  = (const float*)d_in[1];
    const float* b_om  = (const float*)d_in[2];
    const float* w_dcn = (const float*)d_in[3];
    const float* b_dcn = (const float*)d_in[4];
    float* out = (float*)d_out;

    // workspace layout (16B aligned):
    //   [0, 4MB)        wgtArr  float[NPIX*16]
    //   [4MB, 8MB)      srcArr  int[NPIX*16]
    //   [8MB, +128KB)   wAp     f16[16*2*64*32]   (chunk-contiguous)
    //   [.., +192KB)    wOMp    float[16*64*48]
    char* ws = (char*)d_ws;
    float*    wgtArr = (float*)(ws);
    int*      srcArr = (int*)(ws + ((size_t)4 << 20));
    _Float16* wAp    = (_Float16*)(ws + ((size_t)8 << 20));
    float*    wOMp   = (float*)(ws + ((size_t)8 << 20) + ((size_t)128 << 10));

    pack_wdcn_kernel<<<256, 256, 0, stream>>>(w_dcn, wAp);
    pack_wom_kernel <<<192, 256, 0, stream>>>(w_om, wOMp);
    dcn_offset_kernel<<<dim3(128, 4), 64, 0, stream>>>(x, wOMp, b_om, wgtArr, srcArr);
    dcn_gemm_kernel<<<NPIX / 64, 256, 0, stream>>>(x, wAp, b_dcn, wgtArr, srcArr, out);
}
